// TFLiteISTFT_31473520345492
// MI455X (gfx1250) — matrix operand order repaired and checked by reading, not compiled, not run on hardware
//
#include <hip/hip_runtime.h>
#include <hip/hip_bf16.h>

// ---------------- problem constants (match setup_inputs) ----------------
#define BATCH   16
#define TFRM    2048          // frames per batch
#define FFT     1024          // fft_length == frame_length
#define HOP     256
#define NBINS   513           // FFT/2 + 1
#define KDIM    1026          // 2*NBINS (real rows + imag rows)
#define KPAD    1056          // 33 * 32  (zero padded)
#define NKCH    33            // K chunks of 32
#define NTILES  64            // N tiles of 16 (FFT/16)
#define MROWS   (BATCH*TFRM)  // 32768
#define BLOCK_M 32            // rows of A per workgroup
#define ROWSTR  1064          // LDS row stride in f16 (1056 + 8 pad, 16B multiple)
#define OUTLEN  ((TFRM-1)*HOP + FFT)   // 525056 per batch

typedef _Float16 v16h __attribute__((ext_vector_type(16)));
typedef _Float16 v8h  __attribute__((ext_vector_type(8)));
typedef float    v8f  __attribute__((ext_vector_type(8)));

union AFrag { v16h v; v8h h[2]; };
union CFrag { v8f  v; float f[8]; };

// ---------------------------------------------------------------------------
// Kernel 1: build combined DFT basis in WMMA B-fragment order.
//   B[k][n]:  k<513  -> w_f *  cos(2*pi*n*k/FFT) * synwin(n) / FFT
//             k<1026 -> w_f * -sin(2*pi*n*f/FFT) * synwin(n) / FFT   (f=k-513)
//             else   -> 0   (K padding)
//   w_f = 1 for f==0 or f==512 else 2  (Hermitian mirror folded in).
//   B-fragment layout (16-bit B 32x16): lane L holds column N=L&15,
//   K = 8*(L>>4) + (e&7) + 16*(e>>3)  for element e of the lane's v16h.
//   Flat index: (((nTile*33 + kChunk)*32 + lane)*16 + e)
// ---------------------------------------------------------------------------
__global__ __launch_bounds__(256)
void istft_bprep_kernel(_Float16* __restrict__ bfrag) {
    int idx = blockIdx.x * 256 + threadIdx.x;
    const int total = NTILES * NKCH * 32 * 16;
    if (idx >= total) return;
    int e      = idx & 15;
    int lane   = (idx >> 4) & 31;
    int rest   = idx >> 9;
    int kChunk = rest % NKCH;
    int nTile  = rest / NKCH;

    int n = nTile * 16 + (lane & 15);
    int k = kChunk * 32 + ((lane & 16) ? 8 : 0) + (e & 7) + ((e & 8) ? 16 : 0);

    float val = 0.0f;
    if (k < KDIM) {
        int   f  = (k < NBINS) ? k : (k - NBINS);
        float wf = (f == 0 || f == NBINS - 1) ? 1.0f : 2.0f;
        // exact integer phase reduction: (n*f) mod FFT
        int   ph  = (n * f) & (FFT - 1);
        float ang = (2.0f * 3.14159265358979323846f / FFT) * (float)ph;
        float trig = (k < NBINS) ? __cosf(ang) : -__sinf(ang);
        // Hann window / normalization, folded with 1/FFT
        float wang = (2.0f * 3.14159265358979323846f / FFT) * (float)n;
        float win  = 0.5f - 0.5f * __cosf(wang);
        float syn  = win * sqrtf(1.0f - (float)HOP / (float)FFT + 1e-8f);
        val = wf * trig * syn * (1.0f / (float)FFT);
    }
    bfrag[idx] = (_Float16)val;
}

// ---------------------------------------------------------------------------
// Kernel 2: WMMA GEMM.  frames[m][n] = sum_k A[m][k] * B[k][n]
//   A[m][k]: k<513 -> stfts[m][k].real ; k<1026 -> stfts[m][k-513].imag ; 0.
//   Workgroup: 256 threads (8 waves), owns 32 rows of A staged once in LDS
//   as f16 (read from HBM exactly once), then sweeps all N=1024.
//   Wave tiling: mGroup = wave&1 (16 rows), nGroup = wave>>1 (64 cols),
//   4 outer N iterations x 4 accumulators -> full 32x1024 block.
// ---------------------------------------------------------------------------
__global__ __launch_bounds__(256)
void istft_wmma_gemm_kernel(const float2* __restrict__ stfts,
                            const _Float16* __restrict__ bfrag,
                            float* __restrict__ frames) {
    __shared__ __align__(16) _Float16 sA[BLOCK_M * ROWSTR];

    const int tid    = threadIdx.x;
    const int lane   = tid & 31;
    const int wave   = tid >> 5;
    const int mGroup = wave & 1;      // 0..1 : which 16-row half
    const int nGroup = wave >> 1;     // 0..3 : which 64-col strip
    const int blockM = blockIdx.x * BLOCK_M;

    // ---- stage A block into LDS as f16 (real -> k, imag -> k+513) ----
    for (int r = 0; r < BLOCK_M; ++r) {
        const float2* src = stfts + (size_t)(blockM + r) * NBINS;
        _Float16* dst = sA + r * ROWSTR;
        for (int f = tid; f < NBINS; f += 256) {
            float2 v = src[f];
            dst[f]         = (_Float16)v.x;
            dst[NBINS + f] = (_Float16)v.y;
        }
        // zero K padding [1026, 1056)
        if (tid < KPAD - KDIM) dst[KDIM + tid] = (_Float16)0.0f;
    }
    __syncthreads();

    // per-lane A fragment base: row = mGroup*16 + (lane&15),
    // K sub-offset 8 for upper half-wave (ISA interleaved A layout:
    // lane<16 holds K {0..7,16..23}, lane>=16 holds K {8..15,24..31})
    const _Float16* aRow =
        sA + (mGroup * 16 + (lane & 15)) * ROWSTR + ((lane & 16) ? 8 : 0);

    for (int nIter = 0; nIter < 4; ++nIter) {
        const int nTileBase = nIter * 16 + nGroup * 4;
        v8f acc0 = {}, acc1 = {}, acc2 = {}, acc3 = {};

        for (int kc = 0; kc < NKCH; ++kc) {
            AFrag a;
            const v8h* ap = (const v8h*)(aRow + kc * 32);
            a.h[0] = ap[0];   // K bytes [off, off+16)
            a.h[1] = ap[2];   // K bytes [off+32, off+48)

            const v16h* bp =
                (const v16h*)bfrag + ((size_t)nTileBase * NKCH + kc) * 32 + lane;
            v16h b0 = bp[0 * NKCH * 32];
            v16h b1 = bp[1 * NKCH * 32];
            v16h b2 = bp[2 * NKCH * 32];
            v16h b3 = bp[3 * NKCH * 32];

            acc0 = __builtin_amdgcn_wmma_f32_16x16x32_f16(false, a.v, false, b0,
                                                          (short)0, acc0, false, false);
            acc1 = __builtin_amdgcn_wmma_f32_16x16x32_f16(false, a.v, false, b1,
                                                          (short)0, acc1, false, false);
            acc2 = __builtin_amdgcn_wmma_f32_16x16x32_f16(false, a.v, false, b2,
                                                          (short)0, acc2, false, false);
            acc3 = __builtin_amdgcn_wmma_f32_16x16x32_f16(false, a.v, false, b3,
                                                          (short)0, acc3, false, false);
        }

        // store C: lane L -> col = lane&15 (+tile*16), vgpr r -> row 8*(L>=16)+r
        const int rowBase = blockM + mGroup * 16 + ((lane & 16) ? 8 : 0);
        const int colBase = (lane & 15);
        CFrag c[4];
        c[0].v = acc0; c[1].v = acc1; c[2].v = acc2; c[3].v = acc3;
        for (int j = 0; j < 4; ++j) {
            const int col = (nTileBase + j) * 16 + colBase;
            for (int r = 0; r < 8; ++r) {
                frames[(size_t)(rowBase + r) * FFT + col] = c[j].f[r];
            }
        }
    }
}

// ---------------------------------------------------------------------------
// Kernel 3: overlap-add (deterministic fixed-order sum, <=4 frames/sample)
// ---------------------------------------------------------------------------
__global__ __launch_bounds__(256)
void istft_ola_kernel(const float* __restrict__ frames,
                      float* __restrict__ out, int total) {
    int g = blockIdx.x * 256 + threadIdx.x;
    if (g >= total) return;
    int b = g / OUTLEN;
    int s = g - b * OUTLEN;

    int tHi = s >> 8;                       // s / HOP
    if (tHi > TFRM - 1) tHi = TFRM - 1;
    int t0  = s - (FFT - 1) + (HOP - 1);    // ceil((s - (FFT-1)) / HOP)
    int tLo = (t0 > 0) ? (t0 >> 8) : 0;

    float acc = 0.0f;
    const float* fb = frames + (size_t)b * TFRM * FFT;
    for (int t = tLo; t <= tHi; ++t) {
        acc += fb[(size_t)t * FFT + (s - t * HOP)];
    }
    out[g] = acc;
}

// ---------------------------------------------------------------------------
extern "C" void kernel_launch(void* const* d_in, const int* in_sizes, int n_in,
                              void* d_out, int out_size, void* d_ws, size_t ws_size,
                              hipStream_t stream) {
    const float2* stfts = (const float2*)d_in[0];
    float* out = (float*)d_out;

    // workspace layout: [frames f32 : MROWS*FFT] [bfrag f16 : NTILES*NKCH*512]
    float* frames = (float*)d_ws;
    const size_t framesBytes = (size_t)MROWS * FFT * sizeof(float); // 134 MB
    _Float16* bfrag = (_Float16*)((char*)d_ws + framesBytes);       // 2.2 MB

    // 1) basis prep (fragment-ordered, L2 resident afterwards)
    {
        const int total = NTILES * NKCH * 32 * 16;
        istft_bprep_kernel<<<(total + 255) / 256, 256, 0, stream>>>(bfrag);
    }
    // 2) WMMA GEMM: 32 rows per workgroup
    {
        istft_wmma_gemm_kernel<<<MROWS / BLOCK_M, 256, 0, stream>>>(stfts, bfrag, frames);
    }
    // 3) overlap-add
    {
        const int total = BATCH * OUTLEN;   // == out_size
        istft_ola_kernel<<<(total + 255) / 256, 256, 0, stream>>>(frames, out, total);
    }
}
